// GNN_4406636446293
// MI455X (gfx1250) — matrix operand order repaired and checked by reading, not compiled, not run on hardware
//
#include <hip/hip_runtime.h>
#include <hip/hip_bf16.h>

// ---------------------------------------------------------------------------
// Types for CDNA5 WMMA
// ---------------------------------------------------------------------------
typedef __attribute__((ext_vector_type(16))) __bf16 v16bf;
typedef __attribute__((ext_vector_type(2)))  __bf16 v2bf;
typedef __attribute__((ext_vector_type(2)))  float  v2f_t;
typedef __attribute__((ext_vector_type(8)))  float  v8f;

union BfFrag {
    v16bf v;
    uint4 q[2];
    unsigned short u[16];
};

// fp32 -> bf16 via hardware packed convert (v_cvt_pk_bf16_f32)
__device__ __forceinline__ unsigned int pack2bf(float lo, float hi) {
    v2f_t f = {lo, hi};
    v2bf  b = __builtin_convertvector(f, v2bf);
    return __builtin_bit_cast(unsigned int, b);
}

__device__ __forceinline__ unsigned short f2bf(float f) {
    __bf16 b = (__bf16)f;
    return __builtin_bit_cast(unsigned short, b);
}

__device__ __forceinline__ void atomicMaxFloat(float* addr, float v) {
    if (v >= 0.0f) {
        atomicMax((int*)addr, __float_as_int(v));
    } else {
        atomicMin((unsigned int*)addr, __float_as_uint(v));
    }
}

// ---------------------------------------------------------------------------
// Workspace init: hconv <- broadcast(b_conv), segment max <- -inf, sums <- 0
// ---------------------------------------------------------------------------
__global__ void init_ws_kernel(float* __restrict__ hconv,
                               const float* __restrict__ b_conv,
                               float* __restrict__ mseg, float* __restrict__ sseg,
                               float* __restrict__ mb, float* __restrict__ sb,
                               float* __restrict__ pooled, int Nn) {
    long long idx = (long long)blockIdx.x * blockDim.x + threadIdx.x;
    long long tot = (long long)Nn * 512;
    if (idx < tot) hconv[idx] = b_conv[idx & 511];
    if (idx < Nn) { mseg[idx] = -INFINITY; sseg[idx] = 0.0f; }
    if (idx < 8)  { mb[idx]   = -INFINITY; sb[idx]   = 0.0f; }
    if (idx < 8 * 512) pooled[idx] = 0.0f;
}

// ---------------------------------------------------------------------------
// WMMA bf16 GEMM: C[M,N] = A[M,K] * B[K,N] (+ bias[N] if bias != nullptr)
// Block tile 128(M) x 64(N), K-step 32. 256 threads = 8 waves, each wave a
// 32x32 sub-tile = 2x2 v_wmma_f32_16x16x32_bf16.
// Software-pipelined: tile k+1's global loads overlap tile k's WMMAs.
// LDS: A row-major, B transposed; fragments are contiguous b128 LDS loads.
// ---------------------------------------------------------------------------
#define TILE_M 128
#define TILE_N 64
#define TILE_K 32
#define LDSTR  56   // bf16 elements: 112B row stride, 16B aligned, odd bank step

__global__ __launch_bounds__(256) void gemm_wmma_bf16_kernel(
    const float* __restrict__ A, const float* __restrict__ B,
    const float* __restrict__ bias, float* __restrict__ C,
    int M, int N, int K) {
    __shared__ __align__(16) unsigned short lA[TILE_M * LDSTR];
    __shared__ __align__(16) unsigned short lBt[TILE_N * LDSTR];

    const int tid  = threadIdx.x;
    const int lane = tid & 31;
    const int wave = tid >> 5;
    const int wm   = (wave & 3) << 5;   // wave M offset within block tile
    const int wn   = (wave >> 2) << 5;  // wave N offset within block tile
    const int bm   = blockIdx.y * TILE_M;
    const int bn   = blockIdx.x * TILE_N;
    const int half = lane >> 4;
    const int l16  = lane & 15;

    // ---- hoisted staging addresses (rows are K-loop invariant) ----
    const float* aptr[8];
    unsigned int ldsA[8];
    {
        int pc = (tid & 15) * 2;
        int r0 = tid >> 4;
#pragma unroll
        for (int t = 0; t < 8; ++t) {
            int r  = r0 + t * 16;
            int gr = bm + r;
            gr = (gr < M) ? gr : (M - 1);   // clamped rows never stored
            aptr[t] = A + (size_t)gr * K + pc;
            ldsA[t] = r * LDSTR + pc;
        }
    }
    const float* bptr[4];
    unsigned int ldsB[4];
    {
        int pc = (tid & 31) * 2;
        int r0 = tid >> 5;
#pragma unroll
        for (int t = 0; t < 4; ++t) {
            int r = r0 + t * 8;
            bptr[t] = B + (size_t)r * N + bn + pc;
            ldsB[t] = pc * LDSTR + r;
        }
    }
    // fragment read addresses (K-loop invariant)
    const unsigned short* apA[2];
    const unsigned short* apB[2];
#pragma unroll
    for (int i = 0; i < 2; ++i)
        apA[i] = &lA[(wm + i * 16 + l16) * LDSTR + half * 8];
#pragma unroll
    for (int j = 0; j < 2; ++j)
        apB[j] = &lBt[(wn + j * 16 + l16) * LDSTR + half * 8];

    v8f acc[2][2];
#pragma unroll
    for (int i = 0; i < 2; ++i)
#pragma unroll
        for (int j = 0; j < 2; ++j)
            acc[i][j] = (v8f){0.f,0.f,0.f,0.f,0.f,0.f,0.f,0.f};

    // ---- preload K-tile 0 into registers ----
    float2 ra[8], rb[4];
#pragma unroll
    for (int t = 0; t < 8; ++t) ra[t] = *(const float2*)(aptr[t]);
#pragma unroll
    for (int t = 0; t < 4; ++t) rb[t] = *(const float2*)(bptr[t]);

    for (int k0 = 0; k0 < K; k0 += TILE_K) {
        // ---- convert + store held registers to LDS ----
#pragma unroll
        for (int t = 0; t < 8; ++t)
            *(unsigned int*)&lA[ldsA[t]] = pack2bf(ra[t].x, ra[t].y);
#pragma unroll
        for (int t = 0; t < 4; ++t) {
            lBt[ldsB[t]]         = f2bf(rb[t].x);
            lBt[ldsB[t] + LDSTR] = f2bf(rb[t].y);
        }
        __syncthreads();

        // ---- issue next tile's loads (overlap with WMMA below) ----
        int k1 = k0 + TILE_K;
        if (k1 < K) {
#pragma unroll
            for (int t = 0; t < 8; ++t)
                ra[t] = *(const float2*)(aptr[t] + k1);
#pragma unroll
            for (int t = 0; t < 4; ++t)
                rb[t] = *(const float2*)(bptr[t] + (size_t)k1 * N);
        }

        // ---- fragments: contiguous b128 LDS loads ----
        BfFrag af[2], bfg[2];
#pragma unroll
        for (int i = 0; i < 2; ++i) {
            af[i].q[0] = *(const uint4*)(apA[i]);       // K = half*8 .. +7
            af[i].q[1] = *(const uint4*)(apA[i] + 16);  // K = 16+half*8 .. +7
        }
#pragma unroll
        for (int j = 0; j < 2; ++j) {
            bfg[j].q[0] = *(const uint4*)(apB[j]);       // K = half*8 .. +7
            bfg[j].q[1] = *(const uint4*)(apB[j] + 16);  // K = 16+half*8 .. +7
        }
#pragma unroll
        for (int i = 0; i < 2; ++i)
#pragma unroll
            for (int j = 0; j < 2; ++j)
                acc[i][j] = __builtin_amdgcn_wmma_f32_16x16x32_bf16(
                    false, af[i].v, false, bfg[j].v,
                    (short)0, acc[i][j], false, false);
        __syncthreads();
    }

    // ---- write C (+bias): lane l -> col l%16, vgpr r -> row r+8*(l/16)
#pragma unroll
    for (int i = 0; i < 2; ++i) {
#pragma unroll
        for (int j = 0; j < 2; ++j) {
            int coln = bn + wn + j * 16 + l16;
            float bv = bias ? bias[coln] : 0.0f;
#pragma unroll
            for (int r = 0; r < 8; ++r) {
                int rowm = bm + wm + i * 16 + r + half * 8;
                if (rowm < M) C[(size_t)rowm * N + coln] = acc[i][j][r] + bv;
            }
        }
    }
}

// ---------------------------------------------------------------------------
// One wave per row: two simultaneous 512-dots (a_s = h.att_src, a_d = h.att_dst)
// ---------------------------------------------------------------------------
__global__ void rowdot2_kernel(const float* __restrict__ H,
                               const float* __restrict__ v1,
                               const float* __restrict__ v2,
                               float* __restrict__ o1, float* __restrict__ o2,
                               int M, int K) {
    int row  = (int)((blockIdx.x * (long long)blockDim.x + threadIdx.x) >> 5);
    int lane = threadIdx.x & 31;
    if (row >= M) return;
    const float4* hr = (const float4*)(H + (size_t)row * K);
    const float4* w1 = (const float4*)v1;
    const float4* w2 = (const float4*)v2;
    float s1 = 0.f, s2 = 0.f;
    for (int c = lane; c < (K >> 2); c += 32) {
        float4 h = hr[c], a = w1[c], b = w2[c];
        s1 += h.x * a.x + h.y * a.y + h.z * a.z + h.w * a.w;
        s2 += h.x * b.x + h.y * b.y + h.z * b.z + h.w * b.w;
    }
    for (int off = 16; off > 0; off >>= 1) {
        s1 += __shfl_down(s1, off, 32);
        s2 += __shfl_down(s2, off, 32);
    }
    if (lane == 0) { o1[row] = s1; o2[row] = s2; }
}

// One wave per row: single 512-dot + scalar bias (gate = hconv@Wg + bg)
__global__ void rowdot1_kernel(const float* __restrict__ H,
                               const float* __restrict__ v,
                               const float* __restrict__ bias,
                               float* __restrict__ o, int M, int K) {
    int row  = (int)((blockIdx.x * (long long)blockDim.x + threadIdx.x) >> 5);
    int lane = threadIdx.x & 31;
    if (row >= M) return;
    const float4* hr = (const float4*)(H + (size_t)row * K);
    const float4* w  = (const float4*)v;
    float s = 0.f;
    for (int c = lane; c < (K >> 2); c += 32) {
        float4 h = hr[c], a = w[c];
        s += h.x * a.x + h.y * a.y + h.z * a.z + h.w * a.w;
    }
    for (int off = 16; off > 0; off >>= 1) s += __shfl_down(s, off, 32);
    if (lane == 0) o[row] = s + bias[0];
}

// ---------------------------------------------------------------------------
// Edge passes: i < E -> (src[i], dst[i]); i >= E -> self loop (i-E, i-E)
// ---------------------------------------------------------------------------
__device__ __forceinline__ void edge_sd(int i, const int* ei, int E, int& s, int& d) {
    if (i < E) { s = ei[i]; d = ei[E + i]; } else { s = d = i - E; }
}

__device__ __forceinline__ float leaky02(float e) {
    return (e > 0.0f) ? e : 0.2f * e;
}

__global__ void edge_max_kernel(const int* __restrict__ ei,
                                const float* __restrict__ as,
                                const float* __restrict__ ad,
                                float* __restrict__ mseg, int E, int Nn) {
    int i = blockIdx.x * blockDim.x + threadIdx.x;
    if (i >= E + Nn) return;
    int s, d; edge_sd(i, ei, E, s, d);
    atomicMaxFloat(&mseg[d], leaky02(as[s] + ad[d]));
}

__global__ void edge_expsum_kernel(const int* __restrict__ ei,
                                   const float* __restrict__ as,
                                   const float* __restrict__ ad,
                                   const float* __restrict__ mseg,
                                   float* __restrict__ exarr,
                                   float* __restrict__ sseg, int E, int Nn) {
    int i = blockIdx.x * blockDim.x + threadIdx.x;
    if (i >= E + Nn) return;
    int s, d; edge_sd(i, ei, E, s, d);
    float m = mseg[d];
    if (m < -3.0e38f) m = 0.0f;   // isfinite(m) ? m : 0
    float ex = __expf(leaky02(as[s] + ad[d]) - m);
    exarr[i] = ex;
    atomicAdd(&sseg[d], ex);
}

// one wave per edge: hconv[d,:] += (ex/s) * h[src,:]  (float4 gathers)
__global__ void edge_scatter_kernel(const int* __restrict__ ei,
                                    const float* __restrict__ exarr,
                                    const float* __restrict__ sseg,
                                    const float* __restrict__ H,
                                    float* __restrict__ hconv, int E, int Nn) {
    int i    = (int)((blockIdx.x * (long long)blockDim.x + threadIdx.x) >> 5);
    int lane = threadIdx.x & 31;
    if (i >= E + Nn) return;
    int s, d; edge_sd(i, ei, E, s, d);
    float coef = exarr[i] / (sseg[d] + 1e-16f);
    const float4* hs = (const float4*)(H + (size_t)s * 512);
    float* hd = hconv + (size_t)d * 512;
#pragma unroll
    for (int t = 0; t < 4; ++t) {
        int c = t * 32 + lane;               // float4 index 0..127
        float4 v = hs[c];
        float* p = hd + c * 4;
        atomicAdd(p + 0, coef * v.x);
        atomicAdd(p + 1, coef * v.y);
        atomicAdd(p + 2, coef * v.z);
        atomicAdd(p + 3, coef * v.w);
    }
}

// ---------------------------------------------------------------------------
// Batch (G=8) softmax over gate + pooled scatter; also emits g output
// ---------------------------------------------------------------------------
__global__ void batch_max_kernel(const int* __restrict__ batch,
                                 const float* __restrict__ gate,
                                 float* __restrict__ mb, int Nn) {
    int i = blockIdx.x * blockDim.x + threadIdx.x;
    if (i >= Nn) return;
    atomicMaxFloat(&mb[batch[i]], gate[i]);
}

__global__ void batch_expsum_kernel(const int* __restrict__ batch,
                                    const float* __restrict__ gate,
                                    const float* __restrict__ mb,
                                    float* __restrict__ gex,
                                    float* __restrict__ sb, int Nn) {
    int i = blockIdx.x * blockDim.x + threadIdx.x;
    if (i >= Nn) return;
    int b = batch[i];
    float m = mb[b];
    if (m < -3.0e38f) m = 0.0f;
    float ex = __expf(gate[i] - m);
    gex[i] = ex;
    atomicAdd(&sb[b], ex);
}

// one wave per node: pooled[b,:] += g * xn[i,:]; g also written to output
__global__ void node_pool_kernel(const int* __restrict__ batch,
                                 const float* __restrict__ gex,
                                 const float* __restrict__ sb,
                                 const float* __restrict__ xn,
                                 float* __restrict__ pooled,
                                 float* __restrict__ g_out, int Nn) {
    int i    = (int)((blockIdx.x * (long long)blockDim.x + threadIdx.x) >> 5);
    int lane = threadIdx.x & 31;
    if (i >= Nn) return;
    int b = batch[i];
    float g = gex[i] / (sb[b] + 1e-16f);
    if (lane == 0) g_out[i] = g;
    const float4* xr = (const float4*)(xn + (size_t)i * 512);
    float* pr = pooled + (size_t)b * 512;
#pragma unroll
    for (int t = 0; t < 4; ++t) {
        int c = t * 32 + lane;
        float4 v = xr[c];
        float* p = pr + c * 4;
        atomicAdd(p + 0, g * v.x);
        atomicAdd(p + 1, g * v.y);
        atomicAdd(p + 2, g * v.z);
        atomicAdd(p + 3, g * v.w);
    }
}

// ---------------------------------------------------------------------------
// Classifier head (single block): out = sigmoid(relu(pooled@W1+b1)@W2+b2)
// ---------------------------------------------------------------------------
__global__ __launch_bounds__(256) void classifier_kernel(
    const float* __restrict__ pooled, const float* __restrict__ W1,
    const float* __restrict__ b1, const float* __restrict__ W2,
    const float* __restrict__ b2, float* __restrict__ out) {
    __shared__ float t[8 * 512];
    int tid = threadIdx.x;
    for (int o = tid; o < 8 * 512; o += 256) {
        int g = o >> 9, c = o & 511;
        const float* pr = pooled + (size_t)g * 512;
        float s = b1[c];
        for (int k = 0; k < 512; ++k) s += pr[k] * W1[(size_t)k * 512 + c];
        t[o] = (s > 0.0f) ? s : 0.0f;
    }
    __syncthreads();
    int wave = tid >> 5, lane = tid & 31;
    if (wave < 8) {
        float s = 0.0f;
        for (int k = lane; k < 512; k += 32) s += t[wave * 512 + k] * W2[k];
        for (int off = 16; off > 0; off >>= 1) s += __shfl_down(s, off, 32);
        if (lane == 0) out[wave] = 1.0f / (1.0f + __expf(-(s + b2[0])));
    }
}

// ---------------------------------------------------------------------------
// Host-side orchestration
// ---------------------------------------------------------------------------
extern "C" void kernel_launch(void* const* d_in, const int* in_sizes, int n_in,
                              void* d_out, int out_size, void* d_ws, size_t ws_size,
                              hipStream_t stream) {
    const float* x       = (const float*)d_in[0];
    const int*   ei      = (const int*)d_in[1];
    const int*   batch   = (const int*)d_in[2];
    const float* W       = (const float*)d_in[3];
    const float* att_src = (const float*)d_in[4];
    const float* att_dst = (const float*)d_in[5];
    const float* b_conv  = (const float*)d_in[6];
    const float* Wg      = (const float*)d_in[7];
    const float* bg      = (const float*)d_in[8];
    const float* Wn      = (const float*)d_in[9];
    const float* bn      = (const float*)d_in[10];
    const float* W1      = (const float*)d_in[11];
    const float* b1      = (const float*)d_in[12];
    const float* W2      = (const float*)d_in[13];
    const float* b2      = (const float*)d_in[14];

    const int D  = 512, HC = 512;
    const int Nn = in_sizes[0] / D;          // 50000
    const int E  = in_sizes[1] / 2;          // 400000
    const int EN = E + Nn;                   // edges + self loops

    float* outv = (float*)d_out;             // out[0..7]
    float* gout = outv + 8;                  // g[0..Nn-1]

    // workspace layout (floats)
    size_t NH = (size_t)Nn * HC;
    float* h      = (float*)d_ws;
    float* hconv  = h + NH;
    float* xn     = hconv + NH;
    float* a_s    = xn + NH;
    float* a_d    = a_s + Nn;
    float* mseg   = a_d + Nn;
    float* sseg   = mseg + Nn;
    float* exarr  = sseg + Nn;               // EN floats
    float* gate   = exarr + EN;
    float* gex    = gate + Nn;
    float* mb     = gex + Nn;                // 8
    float* sb     = mb + 8;                  // 8
    float* pooled = sb + 8;                  // 8*512

    // 1) init accumulators (hconv pre-seeded with b_conv)
    {
        long long tot = (long long)Nn * HC;
        int blocks = (int)((tot + 255) / 256);
        init_ws_kernel<<<blocks, 256, 0, stream>>>(hconv, b_conv, mseg, sseg,
                                                   mb, sb, pooled, Nn);
    }
    // 2) h = x @ W  (no bias; b_conv is applied post-aggregation)
    {
        dim3 grid(HC / TILE_N, (Nn + TILE_M - 1) / TILE_M);
        gemm_wmma_bf16_kernel<<<grid, 256, 0, stream>>>(x, W, nullptr, h, Nn, HC, D);
    }
    // 3) a_s = h.att_src, a_d = h.att_dst
    {
        long long thr = (long long)Nn * 32;
        rowdot2_kernel<<<(int)((thr + 255) / 256), 256, 0, stream>>>(
            h, att_src, att_dst, a_s, a_d, Nn, HC);
    }
    // 4) per-dst max of leaky_relu attention logits
    edge_max_kernel<<<(EN + 255) / 256, 256, 0, stream>>>(ei, a_s, a_d, mseg, E, Nn);
    // 5) exp + per-dst sum
    edge_expsum_kernel<<<(EN + 255) / 256, 256, 0, stream>>>(ei, a_s, a_d, mseg,
                                                             exarr, sseg, E, Nn);
    // 6) weighted scatter-add aggregation into hconv
    {
        long long thr = (long long)EN * 32;
        edge_scatter_kernel<<<(int)((thr + 255) / 256), 256, 0, stream>>>(
            ei, exarr, sseg, h, hconv, E, Nn);
    }
    // 7) xn = hconv @ Wn + bn ; gate = hconv @ Wg + bg
    {
        dim3 grid(HC / TILE_N, (Nn + TILE_M - 1) / TILE_M);
        gemm_wmma_bf16_kernel<<<grid, 256, 0, stream>>>(hconv, Wn, bn, xn, Nn, HC, HC);
        long long thr = (long long)Nn * 32;
        rowdot1_kernel<<<(int)((thr + 255) / 256), 256, 0, stream>>>(
            hconv, Wg, bg, gate, Nn, HC);
    }
    // 8) batch softmax over gate (G=8)
    batch_max_kernel<<<(Nn + 255) / 256, 256, 0, stream>>>(batch, gate, mb, Nn);
    batch_expsum_kernel<<<(Nn + 255) / 256, 256, 0, stream>>>(batch, gate, mb,
                                                              gex, sb, Nn);
    // 9) pooled = segment_sum(g * xn); emits g output
    {
        long long thr = (long long)Nn * 32;
        node_pool_kernel<<<(int)((thr + 255) / 256), 256, 0, stream>>>(
            batch, gex, sb, xn, pooled, gout, Nn);
    }
    // 10) classifier head -> out[0..7]
    classifier_kernel<<<1, 256, 0, stream>>>(pooled, W1, b1, W2, b2, outv);
}
